// DGCNN_42236708389564
// MI455X (gfx1250) — compile-verified, matrix operand order checked by reading, not run on hardware
//
#include <hip/hip_runtime.h>
#include <hip/hip_bf16.h>

// ---------------- problem constants ----------------
#define BB 8
#define NN 4096
#define BN (BB * NN)
#define KNN 20
#define NCLS 40

typedef __attribute__((ext_vector_type(16))) _Float16 v16h;
typedef __attribute__((ext_vector_type(8)))  _Float16 v8h;
typedef __attribute__((ext_vector_type(8)))  float    v8f;

// ---------------- WMMA helpers ----------------
// A-fragment (16x32 f16 tile, row-major src, stride ld halves).
// Per CDNA5 ISA layout: lane L (0..15) holds row M=L, K in {0..7,16..23};
// lane L+16 holds row M=L, K in {8..15,24..31}.  Two contiguous 16B loads.
__device__ inline v16h load_frag(const _Float16* base, int ld) {
  const int lane = threadIdx.x & 31;
  const int half = lane >> 4;
  const int m    = lane & 15;
  const _Float16* p = base + (size_t)m * ld + half * 8;
  v8h lo = *reinterpret_cast<const v8h*>(p);
  v8h hi = *reinterpret_cast<const v8h*>(p + 16);
  v16h r;
#pragma unroll
  for (int i = 0; i < 8; ++i) { r[i] = lo[i]; r[i + 8] = hi[i]; }
  return r;
}

__device__ inline v8f wmma16x16x32(v16h a, v16h b, v8f c) {
  return __builtin_amdgcn_wmma_f32_16x16x32_f16(
      /*neg_a=*/false, a, /*neg_b=*/false, b,
      /*c_mod=*/(short)0, c, /*reuse_a=*/false, /*reuse_b=*/false);
}

// ---------------- small prep kernels ----------------
__global__ __launch_bounds__(256) void cvt_f32_f16_kernel(
    const float* __restrict__ src, _Float16* __restrict__ dst, int n) {
  int t = blockIdx.x * 256 + threadIdx.x;
  if (t < n) dst[t] = (_Float16)src[t];
}

// W [I x O] row-major (in x out, torch style)  ->  Wt [O x KP] f16, zero-padded K
__global__ __launch_bounds__(256) void prep_wt_kernel(
    const float* __restrict__ W, _Float16* __restrict__ Wt, int I, int O, int KP) {
  int t = blockIdx.x * 256 + threadIdx.x;
  if (t >= O * KP) return;
  int o = t / KP, k = t % KP;
  Wt[t] = (k < I) ? (_Float16)W[(size_t)k * O + o] : (_Float16)0.f;
}

__global__ __launch_bounds__(256) void zero_i32_kernel(int* __restrict__ p, int n) {
  int t = blockIdx.x * 256 + threadIdx.x;
  if (t < n) p[t] = 0;
}

__global__ __launch_bounds__(256) void sqnorm_kernel(
    const _Float16* __restrict__ X, float* __restrict__ sq, int D) {
  int i = blockIdx.x * 256 + threadIdx.x;
  if (i >= BN) return;
  const _Float16* p = X + (size_t)i * D;
  float s = 0.f;
  for (int c = 0; c < D; ++c) { float v = (float)p[c]; s += v * v; }
  sq[i] = s;
}

// ---------------- kNN on 3-d positions (selection-bound, scalar) ----------------
__global__ __launch_bounds__(256) void knn_pos_kernel(
    const float* __restrict__ pos, int* __restrict__ idx) {
  __shared__ float sp[256 * 3];
  const int gid  = blockIdx.x * 256 + threadIdx.x;   // global point (block within one batch)
  const int b    = gid / NN;
  const int base = b * NN;
  const float x0 = pos[(size_t)gid * 3 + 0];
  const float x1 = pos[(size_t)gid * 3 + 1];
  const float x2 = pos[(size_t)gid * 3 + 2];
  float bd[KNN]; int bi[KNN];
#pragma unroll
  for (int s = 0; s < KNN; ++s) { bd[s] = 1e30f; bi[s] = base; }
  for (int t0 = 0; t0 < NN; t0 += 256) {
    __syncthreads();
    int j = base + t0 + threadIdx.x;
    sp[threadIdx.x * 3 + 0] = pos[(size_t)j * 3 + 0];
    sp[threadIdx.x * 3 + 1] = pos[(size_t)j * 3 + 1];
    sp[threadIdx.x * 3 + 2] = pos[(size_t)j * 3 + 2];
    __syncthreads();
    for (int jj = 0; jj < 256; ++jj) {
      int jg = base + t0 + jj;
      float d0 = x0 - sp[jj * 3 + 0];
      float d1 = x1 - sp[jj * 3 + 1];
      float d2 = x2 - sp[jj * 3 + 2];
      float d = d0 * d0 + d1 * d1 + d2 * d2;
      if (jg != gid && d < bd[KNN - 1]) {
        float cd = d; int ci = jg;
#pragma unroll
        for (int s = 0; s < KNN; ++s) {
          if (cd < bd[s]) { float td = bd[s]; int ti = bi[s];
                            bd[s] = cd; bi[s] = ci; cd = td; ci = ti; }
        }
      }
    }
  }
#pragma unroll
  for (int s = 0; s < KNN; ++s) idx[(size_t)gid * KNN + s] = bi[s];
}

// ---------------- kNN on 64-d features: WMMA Gram strips + parallel top-k ----------------
// 16 query rows per WG. All 8 waves build 16x512 Gram chunks with WMMA; then all
// 256 threads scan (16 threads per row, stride 16), each keeping a private top-20
// in registers; a 20-round min-merge across each row's 16 partial lists finishes.
__global__ __launch_bounds__(256) void knn_feat_kernel(
    const _Float16* __restrict__ X /*[BN x 64]*/, const float* __restrict__ sq,
    int* __restrict__ idx) {
  constexpr int CH = 512;
  constexpr int DS = CH + 8;                 // pad to dodge LDS bank conflicts
  __shared__ __align__(16) float dbuf[16 * DS];
  __shared__ float sqs[CH];
  __shared__ float hd[256];
  __shared__ int   hix[256];
  __shared__ int   winner[16];

  const int tid  = threadIdx.x;
  const int wave = tid >> 5;
  const int lane = tid & 31;
  const int half = lane >> 4, nl = lane & 15;
  const int q0   = blockIdx.x * 16;          // 16 query rows (one batch, N%16==0)
  const int b    = q0 / NN;
  const int base = b * NN;
  const int r    = tid >> 4;                 // query row owned by this thread
  const int lr   = tid & 15;                 // lane-within-row

  // query A-fragments, K=64 -> 2 k-steps
  v16h a0 = load_frag(X + (size_t)q0 * 64 + 0, 64);
  v16h a1 = load_frag(X + (size_t)q0 * 64 + 32, 64);

  const int   ig  = q0 + r;
  const float sqi = sq[ig];
  float bd[KNN]; int bi[KNN];
#pragma unroll
  for (int s = 0; s < KNN; ++s) { bd[s] = 1e30f; bi[s] = base; }

  for (int ch = 0; ch < NN; ch += CH) {
    __syncthreads();                          // prev scan done before overwrite
    for (int t = tid; t < CH; t += 256) sqs[t] = sq[base + ch + t];
    // 32 candidate tiles per chunk, 8 waves
    for (int t = wave; t < CH / 16; t += 8) {
      const size_t j0 = (size_t)(base + ch + t * 16);
      v16h b0 = load_frag(X + j0 * 64 + 0, 64);
      v16h b1 = load_frag(X + j0 * 64 + 32, 64);
      v8f c = {};
      c = wmma16x16x32(a0, b0, c);
      c = wmma16x16x32(a1, b1, c);
#pragma unroll
      for (int v = 0; v < 8; ++v)
        dbuf[(v + 8 * half) * DS + t * 16 + nl] = c[v];
    }
    __syncthreads();
    // parallel scan: 16 threads per row, stride 16
    for (int jj = lr; jj < CH; jj += 16) {
      int jg = base + ch + jj;
      float d = sqi + sqs[jj] - 2.f * dbuf[r * DS + jj];
      if (jg != ig && d < bd[KNN - 1]) {
        float cd = d; int ci = jg;
#pragma unroll
        for (int s = 0; s < KNN; ++s) {
          if (cd < bd[s]) { float td = bd[s]; int ti = bi[s];
                            bd[s] = cd; bi[s] = ci; cd = td; ci = ti; }
        }
      }
    }
  }
  // merge: 20 rounds of min-selection across the 16 partial lists of each row
  __syncthreads();
  for (int s = 0; s < KNN; ++s) {
    hd[tid] = bd[0]; hix[tid] = bi[0];
    __syncthreads();
    if (lr == 0) {
      const int bt = r * 16;
      float best = hd[bt]; int w = 0;
      for (int l = 1; l < 16; ++l) {
        float v = hd[bt + l];
        if (v < best) { best = v; w = l; }
      }
      winner[r] = w;
      idx[(size_t)ig * KNN + s] = hix[bt + w];
    }
    __syncthreads();
    if (lr == winner[r]) {
#pragma unroll
      for (int u = 0; u < KNN - 1; ++u) { bd[u] = bd[u + 1]; bi[u] = bi[u + 1]; }
      bd[KNN - 1] = 1e30f;
    }
  }
}

// ---------------- fused EdgeConv (gather + 3-layer WMMA MLP + max-over-K) ----------------
__device__ inline void mlp_layer(const _Float16* in, int inStride, int KD,
                                 const _Float16* __restrict__ Wt,
                                 const float* __restrict__ bias,
                                 _Float16* out, int outStride, int HO, int MT) {
  const int wave = threadIdx.x >> 5, nwaves = blockDim.x >> 5;
  const int lane = threadIdx.x & 31;
  const int half = lane >> 4, nl = lane & 15;
  const int NT = HO >> 4;
  for (int job = wave; job < MT * NT; job += nwaves) {
    const int mt = job / NT, nt = job % NT;
    v8f c = {};
    for (int k = 0; k < KD; k += 32) {
      v16h a = load_frag(in + (size_t)mt * 16 * inStride + k, inStride);
      v16h b = load_frag(Wt + (size_t)nt * 16 * KD + k, KD);
      c = wmma16x16x32(a, b, c);
    }
    const float bb = bias[nt * 16 + nl];
#pragma unroll
    for (int v = 0; v < 8; ++v) {
      int row = mt * 16 + v + 8 * half;
      float x = c[v] + bb;
      x = x > 0.f ? x : 0.f;
      out[(size_t)row * outStride + nt * 16 + nl] = (_Float16)x;
    }
  }
}

__device__ inline void mlp_layer_max(const _Float16* in, int inStride, int KD,
                                     const _Float16* __restrict__ Wt,
                                     const float* __restrict__ bias,
                                     int* acc, int HO, int MT) {
  const int wave = threadIdx.x >> 5, nwaves = blockDim.x >> 5;
  const int lane = threadIdx.x & 31;
  const int half = lane >> 4, nl = lane & 15;
  const int NT = HO >> 4;
  for (int job = wave; job < MT * NT; job += nwaves) {
    const int mt = job / NT, nt = job % NT;
    v8f c = {};
    for (int k = 0; k < KD; k += 32) {
      v16h a = load_frag(in + (size_t)mt * 16 * inStride + k, inStride);
      v16h b = load_frag(Wt + (size_t)nt * 16 * KD + k, KD);
      c = wmma16x16x32(a, b, c);
    }
    const float bb = bias[nt * 16 + nl];
#pragma unroll
    for (int v = 0; v < 8; ++v) {
      int row = mt * 16 + v + 8 * half;       // local edge row
      int pt  = row / KNN;                    // local point
      float x = c[v] + bb;
      x = x > 0.f ? x : 0.f;                  // relu >= 0 -> int-bit ordering == float ordering
      atomicMax(&acc[pt * HO + nt * 16 + nl], __float_as_int(x));
    }
  }
}

template <int PTS, int DF, int KP0, int H1, int H2, int HOUT>
__global__ __launch_bounds__(256) void edgeconv_kernel(
    const _Float16* __restrict__ X /*[BN x DF] f16*/, const int* __restrict__ idx,
    const _Float16* __restrict__ W1t, const float* __restrict__ b1,
    const _Float16* __restrict__ W2t, const float* __restrict__ b2,
    const _Float16* __restrict__ W3t, const float* __restrict__ b3,
    _Float16* __restrict__ OUTH /*[BN x HOUT] f16*/) {
  constexpr int E  = PTS * KNN;
  constexpr int MT = E / 16;
  constexpr int WA = (KP0 > H2) ? KP0 : H2;
  __shared__ __align__(16) _Float16 bufA[E * WA];
  __shared__ __align__(16) _Float16 bufB[E * H1];
  __shared__ int acc[PTS * HOUT];

  const int tid = threadIdx.x;
  const int p0  = blockIdx.x * PTS;

  for (int t = tid; t < PTS * HOUT; t += 256) acc[t] = 0;

  // gather h0 = [x_i, x_j - x_i] (zero pad to WA cols)
  if constexpr ((DF % 8) == 0 && (2 * DF) == WA) {
    // vectorized: one 16B v8h per iteration, packed f16 subtract
    constexpr int VW = WA / 8;
    for (int t = tid; t < E * VW; t += 256) {
      const int e = t / VW, cv = t % VW;
      const int c = cv * 8;
      const int p = p0 + e / KNN;
      v8h val;
      if (c < DF) {
        val = *reinterpret_cast<const v8h*>(X + (size_t)p * DF + c);
      } else {
        const int j  = idx[(size_t)p * KNN + (e % KNN)];
        const int cc = c - DF;
        v8h xj = *reinterpret_cast<const v8h*>(X + (size_t)j * DF + cc);
        v8h xi = *reinterpret_cast<const v8h*>(X + (size_t)p * DF + cc);
        val = xj - xi;
      }
      *reinterpret_cast<v8h*>(bufA + (size_t)e * WA + c) = val;
    }
  } else {
    for (int t = tid; t < E * WA; t += 256) {
      const int e = t / WA, c = t % WA;
      _Float16 v = (_Float16)0.f;
      if (c < 2 * DF) {
        const int p = p0 + e / KNN;
        const int j = idx[(size_t)p * KNN + (e % KNN)];
        if (c < DF) {
          v = X[(size_t)p * DF + c];
        } else {
          const int cc = c - DF;
          v = (_Float16)((float)X[(size_t)j * DF + cc] - (float)X[(size_t)p * DF + cc]);
        }
      }
      bufA[t] = v;
    }
  }
  __syncthreads();
  mlp_layer(bufA, WA, KP0, W1t, b1, bufB, H1, H1, MT);
  __syncthreads();
  mlp_layer(bufB, H1, H1, W2t, b2, bufA, WA, H2, MT);
  __syncthreads();
  mlp_layer_max(bufA, WA, H2, W3t, b3, acc, HOUT, MT);
  __syncthreads();
  for (int t = tid; t < PTS * HOUT; t += 256) {
    float v = __int_as_float(acc[t]);
    OUTH[(size_t)(p0 + t / HOUT) * HOUT + (t % HOUT)] = (_Float16)v;
  }
}

// ---------------- lin0 (WMMA) fused with per-batch segment-max pool ----------------
__global__ __launch_bounds__(256) void lin0_pool_kernel(
    const _Float16* __restrict__ X /*[BN x 256]*/,
    const _Float16* __restrict__ Wt /*[512 x 256]*/,
    const float* __restrict__ bias, int* __restrict__ pooled /*[B x 512] bits*/) {
  const int wave = threadIdx.x >> 5;
  const int lane = threadIdx.x & 31;
  const int half = lane >> 4, nl = lane & 15;
  const int r0 = blockIdx.x * 16;            // 16 rows, same batch (N%16==0)
  const int b  = r0 / NN;
  // hoist A fragments (reused for 4 N-tiles per wave)
  v16h a[8];
#pragma unroll
  for (int kk = 0; kk < 8; ++kk)
    a[kk] = load_frag(X + (size_t)r0 * 256 + kk * 32, 256);
  for (int nt = wave; nt < 32; nt += 8) {
    v8f c = {};
#pragma unroll
    for (int kk = 0; kk < 8; ++kk) {
      v16h bf = load_frag(Wt + (size_t)nt * 16 * 256 + kk * 32, 256);
      c = wmma16x16x32(a[kk], bf, c);
    }
    const float bb = bias[nt * 16 + nl];
#pragma unroll
    for (int v = 0; v < 8; ++v) {
      float x = c[v] + bb;
      x = x > 0.f ? x : 0.f;
      atomicMax(&pooled[b * 512 + nt * 16 + nl], __float_as_int(x));
    }
  }
}

// ---------------- classification head (tiny, one block per batch row) ----------------
__global__ __launch_bounds__(256) void head_kernel(
    const int* __restrict__ pooled,
    const float* __restrict__ W1, const float* __restrict__ b1,   // 512x256
    const float* __restrict__ W2, const float* __restrict__ b2,   // 256x256
    const float* __restrict__ W3, const float* __restrict__ b3,   // 256x40
    float* __restrict__ out) {
  __shared__ float g[512], y1[256], y2[256], y3[NCLS], red[2];
  const int b = blockIdx.x, tid = threadIdx.x;
  for (int t = tid; t < 512; t += 256) g[t] = __int_as_float(pooled[b * 512 + t]);
  __syncthreads();
  {
    float s = b1[tid];
    for (int k = 0; k < 512; ++k) s += g[k] * W1[(size_t)k * 256 + tid];
    y1[tid] = s > 0.f ? s : 0.f;
  }
  __syncthreads();
  {
    float s = b2[tid];
    for (int k = 0; k < 256; ++k) s += y1[k] * W2[(size_t)k * 256 + tid];
    y2[tid] = s > 0.f ? s : 0.f;
  }
  __syncthreads();
  if (tid < NCLS) {
    float s = b3[tid];
    for (int k = 0; k < 256; ++k) s += y2[k] * W3[(size_t)k * NCLS + tid];
    y3[tid] = s;
  }
  __syncthreads();
  if (tid == 0) {
    float mx = -1e30f;
    for (int i = 0; i < NCLS; ++i) mx = y3[i] > mx ? y3[i] : mx;
    float se = 0.f;
    for (int i = 0; i < NCLS; ++i) se += __expf(y3[i] - mx);
    red[0] = mx; red[1] = __logf(se);
  }
  __syncthreads();
  if (tid < NCLS) out[b * NCLS + tid] = y3[tid] - red[0] - red[1];
}

// ---------------- launch ----------------
extern "C" void kernel_launch(void* const* d_in, const int* in_sizes, int n_in,
                              void* d_out, int out_size, void* d_ws, size_t ws_size,
                              hipStream_t stream) {
  (void)in_sizes; (void)n_in; (void)out_size; (void)ws_size;
  const float* pos   = (const float*)d_in[0];
  const float* m1w1  = (const float*)d_in[2];  const float* m1b1 = (const float*)d_in[3];
  const float* m1w2  = (const float*)d_in[4];  const float* m1b2 = (const float*)d_in[5];
  const float* m1w3  = (const float*)d_in[6];  const float* m1b3 = (const float*)d_in[7];
  const float* m2w1  = (const float*)d_in[8];  const float* m2b1 = (const float*)d_in[9];
  const float* m2w2  = (const float*)d_in[10]; const float* m2b2 = (const float*)d_in[11];
  const float* m2w3  = (const float*)d_in[12]; const float* m2b3 = (const float*)d_in[13];
  const float* l0w   = (const float*)d_in[14]; const float* l0b  = (const float*)d_in[15];
  const float* l1w   = (const float*)d_in[16]; const float* l1b  = (const float*)d_in[17];
  const float* l2w   = (const float*)d_in[18]; const float* l2b  = (const float*)d_in[19];
  const float* l3w   = (const float*)d_in[20]; const float* l3b  = (const float*)d_in[21];
  float* out = (float*)d_out;

  // workspace layout
  char* ws = (char*)d_ws;
  size_t off = 0;
  auto alloc = [&](size_t bytes) { size_t o = off; off = (off + bytes + 255) & ~(size_t)255; return o; };
  _Float16* posh  = (_Float16*)(ws + alloc((size_t)BN * 3 * 2));
  int*      idx1  = (int*)     (ws + alloc((size_t)BN * KNN * 4));
  int*      idx2  = (int*)     (ws + alloc((size_t)BN * KNN * 4));
  _Float16* x1h   = (_Float16*)(ws + alloc((size_t)BN * 64 * 2));
  _Float16* x2h   = (_Float16*)(ws + alloc((size_t)BN * 256 * 2));
  float*    sqb   = (float*)   (ws + alloc((size_t)BN * 4));
  int*      pool  = (int*)     (ws + alloc((size_t)BB * 512 * 4));
  _Float16* w1t_1 = (_Float16*)(ws + alloc((size_t)64 * 32 * 2));
  _Float16* w2t_1 = (_Float16*)(ws + alloc((size_t)64 * 64 * 2));
  _Float16* w3t_1 = (_Float16*)(ws + alloc((size_t)64 * 64 * 2));
  _Float16* w1t_2 = (_Float16*)(ws + alloc((size_t)128 * 128 * 2));
  _Float16* w2t_2 = (_Float16*)(ws + alloc((size_t)128 * 128 * 2));
  _Float16* w3t_2 = (_Float16*)(ws + alloc((size_t)256 * 128 * 2));
  _Float16* w0t   = (_Float16*)(ws + alloc((size_t)512 * 256 * 2));

  // prep: f16 weights (transposed, K-padded) + f16 positions
  cvt_f32_f16_kernel<<<(BN * 3 + 255) / 256, 256, 0, stream>>>(pos, posh, BN * 3);
  prep_wt_kernel<<<(64 * 32 + 255) / 256, 256, 0, stream>>>(m1w1, w1t_1, 6, 64, 32);
  prep_wt_kernel<<<(64 * 64 + 255) / 256, 256, 0, stream>>>(m1w2, w2t_1, 64, 64, 64);
  prep_wt_kernel<<<(64 * 64 + 255) / 256, 256, 0, stream>>>(m1w3, w3t_1, 64, 64, 64);
  prep_wt_kernel<<<(128 * 128 + 255) / 256, 256, 0, stream>>>(m2w1, w1t_2, 128, 128, 128);
  prep_wt_kernel<<<(128 * 128 + 255) / 256, 256, 0, stream>>>(m2w2, w2t_2, 128, 128, 128);
  prep_wt_kernel<<<(256 * 128 + 255) / 256, 256, 0, stream>>>(m2w3, w3t_2, 128, 256, 128);
  prep_wt_kernel<<<(512 * 256 + 255) / 256, 256, 0, stream>>>(l0w, w0t, 256, 512, 256);

  // stage 1: knn on positions, EdgeConv1 (6->64->64->64, max over 20)
  knn_pos_kernel<<<BN / 256, 256, 0, stream>>>(pos, idx1);
  edgeconv_kernel<8, 3, 32, 64, 64, 64><<<BN / 8, 256, 0, stream>>>(
      posh, idx1, w1t_1, m1b1, w2t_1, m1b2, w3t_1, m1b3, x1h);

  // stage 2: knn on 64-d features (WMMA Gram), EdgeConv2 (128->128->128->256)
  sqnorm_kernel<<<BN / 256, 256, 0, stream>>>(x1h, sqb, 64);
  knn_feat_kernel<<<BN / 16, 256, 0, stream>>>(x1h, sqb, idx2);
  edgeconv_kernel<8, 64, 128, 128, 128, 256><<<BN / 8, 256, 0, stream>>>(
      x1h, idx2, w1t_2, m2b1, w2t_2, m2b2, w3t_2, m2b3, x2h);

  // stage 3: lin0 + relu + per-batch max pool (fused), then head MLP + log_softmax
  zero_i32_kernel<<<(BB * 512 + 255) / 256, 256, 0, stream>>>(pool, BB * 512);
  lin0_pool_kernel<<<BN / 16, 256, 0, stream>>>(x2h, w0t, l0b, pool);
  head_kernel<<<BB, 256, 0, stream>>>(pool, l1w, l1b, l2w, l2b, l3w, l3b, out);
}
